// SemanticSyntaxHSGModel_56341380989570
// MI455X (gfx1250) — compile-verified, matrix operand order checked by reading, not run on hardware
//
#include <hip/hip_runtime.h>
#include <hip/hip_bf16.h>
#include <math.h>
#include <stdint.h>

typedef __bf16 bf16;
typedef __attribute__((ext_vector_type(16))) __bf16 v16bf;
typedef __attribute__((ext_vector_type(8)))  float  v8f;

#define NSENT 8192
#define LW    50
#define LP    64      // L padded to WMMA tiles
#define DIN   300
#define DP    320     // D padded to K multiple of 32
#define HMID  150
#define HP    160
#define OOUT  64
#define CROWS 128
#define THRESHV 0.7f

#if defined(__HIP_DEVICE_COMPILE__) && \
    __has_builtin(__builtin_amdgcn_global_load_async_to_lds_b128) && \
    __has_builtin(__builtin_amdgcn_s_wait_asynccnt)
#define HAVE_GFX1250_ASYNC 1
// Parameter types per clang diagnostic: (int4 AS1*, int4 AS3*, imm, imm)
typedef int v4i __attribute__((vector_size(16)));
typedef v4i __attribute__((address_space(1)))* gv4i_p;
typedef v4i __attribute__((address_space(3)))* lv4i_p;
#else
#define HAVE_GFX1250_ASYNC 0
#endif

// ---------------------------------------------------------------- WMMA helpers
__device__ __forceinline__ v8f wmma_bf16(v16bf a, v16bf b, v8f c) {
  return __builtin_amdgcn_wmma_f32_16x16x32_bf16(false, a, false, b, (short)0, c,
                                                 false, false);
}

// 16x32 A-fragment (also B-fragment for D = X * Y^T with Y row-major) from
// row-major bf16 [rows][ld] at (r0, k0). Contiguous 2x16B per lane.
// Lanes 0-15: row=r0+lane, K=0..7,16..23 ; lanes 16-31: K=8..15,24..31.
__device__ __forceinline__ v16bf frag_rowmajor(const bf16* s, int ld, int r0,
                                               int k0, int lane) {
  const bf16* p = s + (size_t)(r0 + (lane & 15)) * ld + k0 + ((lane >> 4) << 3);
  v16bf f;
#pragma unroll
  for (int i = 0; i < 8; ++i) { f[i] = p[i]; f[8 + i] = p[16 + i]; }
  return f;
}

// ------------------------------------------------- weight prep: f32 -> bf16^T
// src: [K][O] row-major f32 ; dst: [Op][Kp] row-major bf16, zero padded.
__global__ void prep_wt(const float* __restrict__ src, bf16* __restrict__ dst,
                        int K, int O, int Kp, int Op) {
  int idx = blockIdx.x * 256 + threadIdx.x;
  if (idx >= Op * Kp) return;
  int o = idx / Kp, k = idx - o * Kp;
  float v = (o < O && k < K) ? src[(size_t)k * O + o] : 0.f;
  dst[idx] = (bf16)v;
}

constexpr unsigned SMEM_SENT =
    (unsigned)((LP * DP + LP * LP + LP * HP + LP * HP) * sizeof(bf16) +
               (LP * OOUT + LP + LP) * sizeof(float) + LP * sizeof(int));
constexpr unsigned SMEM_CLS =
    (unsigned)((CROWS * 192 + CROWS * 128) * sizeof(bf16));

// ------------------------------------------ per-sentence SAGE + attention pool
__global__ __launch_bounds__(256)
void sage_graph_kernel(const int* __restrict__ words, const int* __restrict__ adj,
                       const float* __restrict__ embed,
                       const bf16* __restrict__ W1sT, const bf16* __restrict__ W1nT,
                       const float* __restrict__ b1,
                       const bf16* __restrict__ W2sT, const bf16* __restrict__ W2nT,
                       const float* __restrict__ b2,
                       const float* __restrict__ gw, const float* __restrict__ gb,
                       float* __restrict__ outFeat, int isSem) {
  extern __shared__ unsigned char smem[];
  bf16*  sEmb  = (bf16*)smem;                 // [LP][DP]  raw embeddings
  bf16*  sA    = sEmb + LP * DP;              // [LP][LP]  Afn[v][u]
  bf16*  sYT   = sA + LP * LP;                // [HP][LP]  (x@Wn)^T scratch
  bf16*  sH1   = sYT + LP * HP;               // [LP][HP]  (phase0: f32 staging)
  float* sH2   = (float*)(sH1 + LP * HP);     // [LP][OOUT]
  float* sGate = sH2 + LP * OOUT;             // [LP]
  float* sInv  = sGate + LP;                  // [LP]
  int*   sMask = (int*)(sInv + LP);           // [LP]

  const int tid = threadIdx.x, lane = tid & 31, w = tid >> 5;
  const int sent = blockIdx.x;

  // -- Phase 0: gather embedding rows into LDS, f32 inverse norms, mask.
#if HAVE_GFX1250_ASYNC
  {
    // Async-DMA each 1200B embed row into a per-wave staging slot, then
    // norm-reduce + bf16-pack from LDS. sH1 region is free this early.
    float* stage = (float*)((char*)sH1 + w * 1280);
    for (int r = w; r < LP; r += 8) {
      int wd = (r < LW) ? words[(size_t)sent * LW + r] : 0;
      int valid = wd > 0;
      if (valid) {
        const char* g = (const char*)(embed + (size_t)wd * DIN) + lane * 16;
        char*       l = (char*)stage + lane * 16;
        __builtin_amdgcn_global_load_async_to_lds_b128(
            (gv4i_p)(uintptr_t)g,
            (lv4i_p)(unsigned)(uintptr_t)l, 0, 0);
        __builtin_amdgcn_global_load_async_to_lds_b128(
            (gv4i_p)(uintptr_t)(g + 512),
            (lv4i_p)(unsigned)(uintptr_t)(l + 512), 0, 0);
        if (lane < 11)  // tail: 1200 - 1024 = 176 B = 11 x 16B
          __builtin_amdgcn_global_load_async_to_lds_b128(
              (gv4i_p)(uintptr_t)(g + 1024),
              (lv4i_p)(unsigned)(uintptr_t)(l + 1024), 0, 0);
        __builtin_amdgcn_s_wait_asynccnt(0);
      }
      float ss = 0.f;
      for (int c = lane; c < DP; c += 32) {
        float v = (valid && c < DIN) ? stage[c] : 0.f;
        sEmb[r * DP + c] = (bf16)v;
        ss += v * v;
      }
#pragma unroll
      for (int off = 16; off > 0; off >>= 1) ss += __shfl_xor(ss, off, 32);
      if (lane == 0) {
        sMask[r] = valid;
        sInv[r] = valid ? 1.f / fmaxf(sqrtf(ss), 1e-8f) : 0.f;
      }
    }
  }
#else
  for (int r = w; r < LP; r += 8) {
    int wd = (r < LW) ? words[(size_t)sent * LW + r] : 0;
    int valid = wd > 0;
    const float* src = embed + (size_t)wd * DIN;
    float ss = 0.f;
    for (int c = lane; c < DP; c += 32) {
      float v = (valid && c < DIN) ? src[c] : 0.f;
      sEmb[r * DP + c] = (bf16)v;
      ss += v * v;
    }
#pragma unroll
    for (int off = 16; off > 0; off >>= 1) ss += __shfl_xor(ss, off, 32);
    if (lane == 0) {
      sMask[r] = valid;
      sInv[r] = valid ? 1.f / fmaxf(sqrtf(ss), 1e-8f) : 0.f;
    }
  }
#endif
  __syncthreads();

  // -- Phase 1: adjacency  sA[v][u] = A[u][v]
  if (isSem) {
    // Symmetric Gram matrix: only the 10 upper-triangle 16x16 tiles.
    for (int t = w; t < 10; t += 8) {
      int i = (t < 4) ? 0 : (t < 7) ? 1 : (t < 9) ? 2 : 3;
      int j = t - ((i == 0) ? 0 : (i == 1) ? 3 : (i == 2) ? 5 : 6);
      int m0 = i << 4, n0 = j << 4;
      v8f acc = {};
      for (int k0 = 0; k0 < DP; k0 += 32)
        acc = wmma_bf16(frag_rowmajor(sEmb, DP, m0, k0, lane),
                        frag_rowmajor(sEmb, DP, n0, k0, lane), acc);
      int n = n0 + (lane & 15), hi = (lane >> 4) << 3;
#pragma unroll
      for (int r = 0; r < 8; ++r) {
        int m = m0 + r + hi;
        float cosv = acc[r] * sInv[m] * sInv[n];
        bf16 e = (bf16)((cosv > THRESHV && sMask[m] && sMask[n]) ? 1.f : 0.f);
        sA[n * LP + m] = e;                 // A[u=m][v=n], stored transposed
        if (m0 != n0) sA[m * LP + n] = e;   // mirror (cos is exactly symmetric)
      }
    }
  } else {
    for (int idx = tid; idx < LP * LP; idx += 256) {
      int v = idx / LP, u = idx - v * LP;
      float e = (u < LW && v < LW &&
                 adj[(size_t)sent * LW * LW + u * LW + v]) ? 1.f : 0.f;
      sA[v * LP + u] = (bf16)e;
    }
  }
  __syncthreads();

  // -- Phase 2: normalize rows by clamped in-degree: Afn[v][u] = A[u][v]/deg[v]
  for (int v = tid; v < LP; v += 256) {
    float s = 0.f;
    for (int u = 0; u < LP; ++u) s += (float)sA[v * LP + u];
    float inv = 1.f / fmaxf(s, 1.f);
    for (int u = 0; u < LP; ++u)
      sA[v * LP + u] = (bf16)((float)sA[v * LP + u] * inv);
  }
  __syncthreads();

  // -- Phase 3: y1 = x @ W1n, stored transposed sYT[h][u]  (m@W1n == Afn@y1)
  for (int t = w; t < (LP / 16) * (HP / 16); t += 8) {
    int m0 = (t / (HP / 16)) << 4, n0 = (t % (HP / 16)) << 4;
    v8f acc = {};
    for (int k0 = 0; k0 < DP; k0 += 32)
      acc = wmma_bf16(frag_rowmajor(sEmb, DP, m0, k0, lane),
                      frag_rowmajor(W1nT, DP, n0, k0, lane), acc);
    int n = n0 + (lane & 15), hi = (lane >> 4) << 3;
    bf16* q = sYT + n * LP + m0 + hi;       // 8 contiguous bf16 per lane
#pragma unroll
    for (int r = 0; r < 8; ++r) q[r] = (bf16)acc[r];
  }
  __syncthreads();

  // -- Phase 4: h1 = relu(x@W1s + Afn@y1 + b1)
  for (int t = w; t < (LP / 16) * (HP / 16); t += 8) {
    int m0 = (t / (HP / 16)) << 4, n0 = (t % (HP / 16)) << 4;
    v8f acc = {};
    for (int k0 = 0; k0 < DP; k0 += 32)
      acc = wmma_bf16(frag_rowmajor(sEmb, DP, m0, k0, lane),
                      frag_rowmajor(W1sT, DP, n0, k0, lane), acc);
    for (int k0 = 0; k0 < LP; k0 += 32)   // B[k=u][n=h] = sYT[n][k], contiguous
      acc = wmma_bf16(frag_rowmajor(sA, LP, m0, k0, lane),
                      frag_rowmajor(sYT, LP, n0, k0, lane), acc);
    int n = n0 + (lane & 15), hi = (lane >> 4) << 3;
    float bias = (n < HMID) ? b1[n] : 0.f;
#pragma unroll
    for (int r = 0; r < 8; ++r)
      sH1[(m0 + r + hi) * HP + n] = (bf16)fmaxf(acc[r] + bias, 0.f);
  }
  __syncthreads();

  // -- Phase 5: y2 = h1 @ W2n, stored transposed sYT[o][u]
  for (int t = w; t < (LP / 16) * (OOUT / 16); t += 8) {
    int m0 = (t / (OOUT / 16)) << 4, n0 = (t % (OOUT / 16)) << 4;
    v8f acc = {};
    for (int k0 = 0; k0 < HP; k0 += 32)
      acc = wmma_bf16(frag_rowmajor(sH1, HP, m0, k0, lane),
                      frag_rowmajor(W2nT, HP, n0, k0, lane), acc);
    int n = n0 + (lane & 15), hi = (lane >> 4) << 3;
    bf16* q = sYT + n * LP + m0 + hi;
#pragma unroll
    for (int r = 0; r < 8; ++r) q[r] = (bf16)acc[r];
  }
  __syncthreads();

  // -- Phase 6: h2 = h1@W2s + Afn@y2 + b2   (f32)
  for (int t = w; t < (LP / 16) * (OOUT / 16); t += 8) {
    int m0 = (t / (OOUT / 16)) << 4, n0 = (t % (OOUT / 16)) << 4;
    v8f acc = {};
    for (int k0 = 0; k0 < HP; k0 += 32)
      acc = wmma_bf16(frag_rowmajor(sH1, HP, m0, k0, lane),
                      frag_rowmajor(W2sT, HP, n0, k0, lane), acc);
    for (int k0 = 0; k0 < LP; k0 += 32)
      acc = wmma_bf16(frag_rowmajor(sA, LP, m0, k0, lane),
                      frag_rowmajor(sYT, LP, n0, k0, lane), acc);
    int n = n0 + (lane & 15), hi = (lane >> 4) << 3;
    float bias = b2[n];
#pragma unroll
    for (int r = 0; r < 8; ++r) sH2[(m0 + r + hi) * OOUT + n] = acc[r] + bias;
  }
  __syncthreads();

  // -- Phase 7: gated attention pooling (masked softmax over nodes)
  for (int r = tid; r < LP; r += 256) {
    float g = gb[0];
    for (int o = 0; o < OOUT; ++o) g += sH2[r * OOUT + o] * gw[o];
    sGate[r] = sMask[r] ? g : -1e9f;
  }
  __syncthreads();
  float mx = -3.4e38f;
  for (int j = 0; j < LP; ++j) mx = fmaxf(mx, sGate[j]);
  __syncthreads();
  if (tid < LP) sGate[tid] = __expf(sGate[tid] - mx);
  __syncthreads();
  if (tid < OOUT) {
    float ssum = 0.f, o = 0.f;
    for (int j = 0; j < LP; ++j) ssum += sGate[j];
    for (int j = 0; j < LP; ++j) o += sGate[j] * sH2[j * OOUT + tid];
    outFeat[(size_t)sent * OOUT + tid] = o / ssum;
  }
}

// ------------------------------------------------------- classifier (WMMA MLP)
__global__ __launch_bounds__(256)
void classifier_kernel(const float* __restrict__ semFeat,
                       const float* __restrict__ synFeat,
                       const float* __restrict__ hsg,
                       const bf16* __restrict__ W1T, const float* __restrict__ b1,
                       const bf16* __restrict__ W2T, const float* __restrict__ b2,
                       float* __restrict__ out) {
  extern __shared__ unsigned char smem[];
  bf16* sF = (bf16*)smem;          // [CROWS][192]
  bf16* sHid = sF + CROWS * 192;   // [CROWS][128]
  const int tid = threadIdx.x, lane = tid & 31, w = tid >> 5;
  const int row0 = blockIdx.x * CROWS;

  for (int idx = tid; idx < CROWS * 192; idx += 256) {
    int r = idx / 192, c = idx - r * 192;
    size_t gr = (size_t)(row0 + r);
    float v = (c < 64)    ? semFeat[gr * 64 + c]
              : (c < 128) ? synFeat[gr * 64 + (c - 64)]
                          : hsg[gr * 64 + (c - 128)];
    sF[idx] = (bf16)v;
  }
  __syncthreads();

  for (int t = w; t < 64; t += 8) {  // hid = relu(feat @ W1 + b1)
    int m0 = (t >> 3) << 4, n0 = (t & 7) << 4;
    v8f acc = {};
    for (int k0 = 0; k0 < 192; k0 += 32)
      acc = wmma_bf16(frag_rowmajor(sF, 192, m0, k0, lane),
                      frag_rowmajor(W1T, 192, n0, k0, lane), acc);
    int n = n0 + (lane & 15), hi = (lane >> 4) << 3;
    float bias = b1[n];
#pragma unroll
    for (int r = 0; r < 8; ++r)
      sHid[(m0 + r + hi) * 128 + n] = (bf16)fmaxf(acc[r] + bias, 0.f);
  }
  __syncthreads();

  {  // logits = hid @ W2 + b2 (N padded to 16, 2 valid cols)
    int m0 = w << 4;
    v8f acc = {};
    for (int k0 = 0; k0 < 128; k0 += 32)
      acc = wmma_bf16(frag_rowmajor(sHid, 128, m0, k0, lane),
                      frag_rowmajor(W2T, 128, 0, k0, lane), acc);
    int n = lane & 15, hi = (lane >> 4) << 3;
    if (n < 2) {
      float bias = b2[n];
#pragma unroll
      for (int r = 0; r < 8; ++r)
        out[(size_t)(row0 + m0 + r + hi) * 2 + n] = acc[r] + bias;
    }
  }
}

// ---------------------------------------------------------------------- launch
extern "C" void kernel_launch(void* const* d_in, const int* in_sizes, int n_in,
                              void* d_out, int out_size, void* d_ws, size_t ws_size,
                              hipStream_t stream) {
  (void)in_sizes; (void)n_in; (void)out_size; (void)ws_size;
  const int*   words  = (const int*)d_in[0];
  const int*   synw   = (const int*)d_in[1];
  const int*   synadj = (const int*)d_in[2];
  const float* hsg    = (const float*)d_in[3];
  const float* embed  = (const float*)d_in[4];
  const float* semW1s = (const float*)d_in[5];
  const float* semW1n = (const float*)d_in[6];
  const float* semb1  = (const float*)d_in[7];
  const float* semW2s = (const float*)d_in[8];
  const float* semW2n = (const float*)d_in[9];
  const float* semb2  = (const float*)d_in[10];
  const float* semgw  = (const float*)d_in[11];
  const float* semgb  = (const float*)d_in[12];
  const float* synW1s = (const float*)d_in[13];
  const float* synW1n = (const float*)d_in[14];
  const float* synb1  = (const float*)d_in[15];
  const float* synW2s = (const float*)d_in[16];
  const float* synW2n = (const float*)d_in[17];
  const float* synb2  = (const float*)d_in[18];
  const float* syngw  = (const float*)d_in[19];
  const float* syngb  = (const float*)d_in[20];
  const float* clsW1  = (const float*)d_in[21];
  const float* clsb1  = (const float*)d_in[22];
  const float* clsW2  = (const float*)d_in[23];
  const float* clsb2  = (const float*)d_in[24];
  float* out = (float*)d_out;

  // workspace carve
  float* semFeat = (float*)d_ws;                       // [N][64]
  float* synFeat = semFeat + (size_t)NSENT * OOUT;     // [N][64]
  bf16* wp = (bf16*)(synFeat + (size_t)NSENT * OOUT);
  bf16 *semW1sT = wp; wp += HP * DP;
  bf16 *semW1nT = wp; wp += HP * DP;
  bf16 *synW1sT = wp; wp += HP * DP;
  bf16 *synW1nT = wp; wp += HP * DP;
  bf16 *semW2sT = wp; wp += OOUT * HP;
  bf16 *semW2nT = wp; wp += OOUT * HP;
  bf16 *synW2sT = wp; wp += OOUT * HP;
  bf16 *synW2nT = wp; wp += OOUT * HP;
  bf16 *clsW1T  = wp; wp += 128 * 192;
  bf16 *clsW2T  = wp; wp += 16 * 128;

  auto prep = [&](const float* s, bf16* d, int K, int O, int Kp, int Op) {
    int n = Kp * Op;
    prep_wt<<<dim3((n + 255) / 256), dim3(256), 0, stream>>>(s, d, K, O, Kp, Op);
  };
  prep(semW1s, semW1sT, DIN, HMID, DP, HP);
  prep(semW1n, semW1nT, DIN, HMID, DP, HP);
  prep(synW1s, synW1sT, DIN, HMID, DP, HP);
  prep(synW1n, synW1nT, DIN, HMID, DP, HP);
  prep(semW2s, semW2sT, HMID, OOUT, HP, OOUT);
  prep(semW2n, semW2nT, HMID, OOUT, HP, OOUT);
  prep(synW2s, synW2sT, HMID, OOUT, HP, OOUT);
  prep(synW2n, synW2nT, HMID, OOUT, HP, OOUT);
  prep(clsW1, clsW1T, 192, 128, 192, 128);
  prep(clsW2, clsW2T, 128, 2, 128, 16);

  (void)hipFuncSetAttribute(reinterpret_cast<const void*>(sage_graph_kernel),
                            hipFuncAttributeMaxDynamicSharedMemorySize,
                            (int)SMEM_SENT);
  (void)hipFuncSetAttribute(reinterpret_cast<const void*>(classifier_kernel),
                            hipFuncAttributeMaxDynamicSharedMemorySize,
                            (int)SMEM_CLS);

  sage_graph_kernel<<<dim3(NSENT), dim3(256), SMEM_SENT, stream>>>(
      words, synadj, embed, semW1sT, semW1nT, semb1, semW2sT, semW2nT, semb2,
      semgw, semgb, semFeat, 1);
  sage_graph_kernel<<<dim3(NSENT), dim3(256), SMEM_SENT, stream>>>(
      synw, synadj, embed, synW1sT, synW1nT, synb1, synW2sT, synW2nT, synb2,
      syngw, syngb, synFeat, 0);
  classifier_kernel<<<dim3(NSENT / CROWS), dim3(256), SMEM_CLS, stream>>>(
      semFeat, synFeat, hsg, clsW1T, clsb1, clsW2T, clsb2, out);
}
